// Multi_input_MambaClassifier_47553877901711
// MI455X (gfx1250) — compile-verified, matrix operand order checked by reading, not run on hardware
//
#include <hip/hip_runtime.h>
#include <hip/hip_bf16.h>

// ---------------------------------------------------------------------------
// Multi-input Mamba classifier for MI455X (gfx1250, wave32, WMMA).
// GEMM-shaped ops (3^3 convs via implicit GEMM, 1x1 convs, mamba projections)
// use V_WMMA_F32_16X16X32_F16 with block-cooperative LDS staging of the
// im2col B tile and a precomputed k-decode table (no divisions in hot loop).
// ---------------------------------------------------------------------------

typedef __attribute__((ext_vector_type(16))) _Float16 v16h;
typedef __attribute__((ext_vector_type(8)))  _Float16 v8h;
typedef __attribute__((ext_vector_type(8)))  float    v8f;

#define LSEQ   1728          // 12^3 sequence length for mamba
#define BN_EPS 1e-5f

__device__ __forceinline__ float bn_apply(float x, float g, float b, float m, float v) {
    return g * (x - m) * rsqrtf(v + BN_EPS) + b;
}
__device__ __forceinline__ float silu_f(float x) { return x / (1.f + __expf(-x)); }

// ---------------------------------------------------------------------------
// Implicit-GEMM conv / GEMM with WMMA.
//   out[m][n] = epilogue( sum_k W[m][k] * im2col(k, n) )
//   KS==3 : 3x3x3 conv, pad 1, stride {1,2}; K = Cin*27
//   KS==1 : 1x1x1 conv / plain GEMM, stride {1,2}; K = Cin
// Block = 128 threads = 4 waves computing a 64(M) x 64(N) macro-tile:
//   wave w owns rows [w*16, w*16+16) and all 64 columns (4 accumulators).
//   B tile (32K x 64N f16) is gathered cooperatively into LDS once per K-step
//   and consumed by all 4 waves via 16B ds loads.
// Epilogue: optional BN (g!=null), optional residual add, optional ReLU.
// ---------------------------------------------------------------------------
template<int KS>
__global__ __launch_bounds__(128)
void wmma_conv(const float* __restrict__ in, const float* __restrict__ W,
               float* __restrict__ out,
               int Cout, int Cin, int S, int T, int stride,
               const float* __restrict__ g,  const float* __restrict__ bb,
               const float* __restrict__ mm, const float* __restrict__ vv,
               const float* __restrict__ res, int relu,
               long inStr, long outStr, long resStr)
{
    const int N   = T * T * T;
    const int K   = Cin * KS * KS * KS;          // always a multiple of 32 here
    const int S3  = S * S * S;
    const int ntN = (N + 63) >> 6;

    const int tmMac = blockIdx.x / ntN;          // 64-row macro tile
    const int tnMac = blockIdx.x % ntN;          // 64-col macro tile
    const int s     = blockIdx.z;

    const float* inp  = in  + (long)s * inStr;
    float*       outp = out + (long)s * outStr;

    // ---- LDS ----
    __shared__ int tabA[(KS == 3) ? 3456 : 1];   // ci*S^3 + kd*S^2 + kh*S + kw
    __shared__ int tabP[(KS == 3) ? 3456 : 1];   // kd | kh<<8 | kw<<16
    __shared__ alignas(16) _Float16 tileB[64 * 40];  // [n][k], stride 40 halves

    if (KS == 3) {
        for (int k = threadIdx.x; k < K; k += 128) {
            int ci = k / 27, r27 = k - ci * 27;
            int kd = r27 / 9, r9 = r27 - kd * 9;
            int kh = r9 / 3,  kw = r9 - kh * 3;
            tabA[k] = ci * S3 + kd * S * S + kh * S + kw;
            tabP[k] = kd | (kh << 8) | (kw << 16);
        }
    }

    // ---- per-thread gather column setup (fixed for whole kernel) ----
    const int tcol = threadIdx.x & 63;           // column this thread stages
    const int krow = (threadIdx.x >> 6) * 16;    // k_local range start (0 / 16)
    const int ncol = tnMac * 64 + tcol;
    const bool nvalid = (ncol < N);
    int odm1 = 0, ohm1 = 0, owm1 = 0;
    long nbase = 0;
    {
        int nn = nvalid ? ncol : 0;
        int od = nn / (T * T);
        int rem = nn - od * T * T;
        int oh = rem / T;
        int ow = rem - oh * T;
        if (KS == 3) {
            odm1 = od * stride - 1;
            ohm1 = oh * stride - 1;
            owm1 = ow * stride - 1;
            nbase = (long)odm1 * S * S + (long)ohm1 * S + owm1;
        } else {
            nbase = (stride == 1) ? nn
                  : ((long)(od * stride) * S + oh * stride) * S + ow * stride;
        }
    }

    // ---- per-wave compute setup ----
    const int wave = threadIdx.x >> 5;
    const int lane = threadIdx.x & 31;
    const int r    = lane & 15;
    const int hi   = lane >> 4;
    const int mrow = tmMac * 64 + wave * 16 + r; // A row this lane loads
    const bool mok = (mrow < Cout);
    const float* wrow = W + (long)(mok ? mrow : 0) * K;

    v8f acc[4] = {};
    for (int kb = 0; kb < K; kb += 32) {
        // ---- stage B tile: tileB[n][kl], kl in [0,32) ----
        __syncthreads();
#pragma unroll
        for (int i = 0; i < 16; ++i) {
            int kl = krow + i;
            int k  = kb + kl;
            float bv = 0.f;
            if (KS == 3) {
                int pk = tabP[k];
                int kd = pk & 255, kh = (pk >> 8) & 255, kw = pk >> 16;
                bool ok = nvalid &&
                          (unsigned)(odm1 + kd) < (unsigned)S &&
                          (unsigned)(ohm1 + kh) < (unsigned)S &&
                          (unsigned)(owm1 + kw) < (unsigned)S;
                if (ok) bv = inp[(long)tabA[k] + nbase];
            } else {
                if (nvalid) bv = inp[(long)k * S3 + nbase];
            }
            tileB[tcol * 40 + kl] = (_Float16)bv;
        }
        __syncthreads();

        // ---- A fragment: two contiguous 8-float runs (vectorized) ----
        float wtmp[16];
        if (mok) {
            const float4* p0 = (const float4*)(wrow + kb + hi * 8);
            const float4* p1 = (const float4*)(wrow + kb + 16 + hi * 8);
            *(float4*)&wtmp[0]  = p0[0];
            *(float4*)&wtmp[4]  = p0[1];
            *(float4*)&wtmp[8]  = p1[0];
            *(float4*)&wtmp[12] = p1[1];
        } else {
#pragma unroll
            for (int j = 0; j < 16; ++j) wtmp[j] = 0.f;
        }
        v16h a;
#pragma unroll
        for (int j = 0; j < 16; ++j) a[j] = (_Float16)wtmp[j];

        if (mok && kb + 32 < K)                       // gfx1250 global_prefetch
            __builtin_prefetch(wrow + kb + 32, 0, 0);

        // ---- 4 WMMAs: one per 16-column sub-tile ----
#pragma unroll
        for (int sub = 0; sub < 4; ++sub) {
            int nl = sub * 16 + r;
            v8h lo  = *(const v8h*)&tileB[nl * 40 + hi * 8];
            v8h hi8 = *(const v8h*)&tileB[nl * 40 + 16 + hi * 8];
            v16h b;
#pragma unroll
            for (int j = 0; j < 8; ++j) { b[j] = lo[j]; b[j + 8] = hi8[j]; }
            acc[sub] = __builtin_amdgcn_wmma_f32_16x16x32_f16(
                false, a, false, b, (short)0, acc[sub], false, false);
        }
    }

    // ---- epilogue.  C/D layout: VGPR i -> M = i + 8*hi, N = lane&15 ----
#pragma unroll
    for (int sub = 0; sub < 4; ++sub) {
        int nn = tnMac * 64 + sub * 16 + r;
#pragma unroll
        for (int i = 0; i < 8; ++i) {
            int m = tmMac * 64 + wave * 16 + hi * 8 + i;
            if (m < Cout && nn < N) {
                float v2 = acc[sub][i];
                if (g)   v2 = bn_apply(v2, g[m], bb[m], mm[m], vv[m]);
                if (res) v2 += res[(long)s * resStr + (long)m * N + nn];
                if (relu) v2 = fmaxf(v2, 0.f);
                outp[(long)m * N + nn] = v2;
            }
        }
    }
}

// ---------------------------------------------------------------------------
// Stem: conv3 1->64, 3^3, stride 2, pad 1, +bias, then BN+ReLU.  K=27 -> VALU.
// ---------------------------------------------------------------------------
__global__ __launch_bounds__(256)
void stem_kernel(const float* __restrict__ x, float* __restrict__ outA,
                 const float* __restrict__ cw, const float* __restrict__ cb,
                 const float* __restrict__ g, const float* __restrict__ b,
                 const float* __restrict__ m, const float* __restrict__ v, int s0)
{
    int sAbs = s0 + blockIdx.z;
    int bidx, chan;
    if (sAbs < 2) { bidx = sAbs; chan = 0; }
    else          { bidx = (sAbs - 2) / 6; chan = (sAbs - 2) % 6 + 1; }
    const float* in = x + ((long)bidx * 7 + chan) * 110592;   // 48^3

    int idx = blockIdx.x * blockDim.x + threadIdx.x;
    if (idx >= 64 * 13824) return;
    int co = idx / 13824;
    int sp = idx - co * 13824;
    int od = sp / 576, rem = sp - od * 576, oh = rem / 24, ow = rem - oh * 24;

    float acc = cb[co];
#pragma unroll
    for (int kd = 0; kd < 3; ++kd) {
        int id = od * 2 - 1 + kd; if ((unsigned)id >= 48u) continue;
#pragma unroll
        for (int kh = 0; kh < 3; ++kh) {
            int ih = oh * 2 - 1 + kh; if ((unsigned)ih >= 48u) continue;
#pragma unroll
            for (int kw = 0; kw < 3; ++kw) {
                int iw = ow * 2 - 1 + kw; if ((unsigned)iw >= 48u) continue;
                acc += cw[co * 27 + kd * 9 + kh * 3 + kw] *
                       in[((long)id * 48 + ih) * 48 + iw];
            }
        }
    }
    float val = bn_apply(acc, g[co], b[co], m[co], v[co]);
    outA[(long)blockIdx.z * 884736 + idx] = fmaxf(val, 0.f);
}

// Depthwise causal conv1d (k=4, left pad 3) + bias + SiLU.  xm = rows 0..255 of xz.
__global__ __launch_bounds__(256)
void dwconv_silu(const float* __restrict__ xz, float* __restrict__ xm2,
                 const float* __restrict__ cw, const float* __restrict__ cb)
{
    long s = blockIdx.z;
    int idx = blockIdx.x * blockDim.x + threadIdx.x;
    if (idx >= 256 * LSEQ) return;
    int c = idx / LSEQ, l = idx - c * LSEQ;
    const float* row = xz + s * 884736 + (long)c * LSEQ;
    float acc = cb[c];
#pragma unroll
    for (int t = 0; t < 4; ++t) {
        int li = l - 3 + t;
        if (li >= 0) acc += cw[c * 4 + t] * row[li];
    }
    xm2[s * 442368 + idx] = silu_f(acc);
}

// dt = softplus(dt_proj @ dt_in + dt_bias).  K=8 -> VALU.
__global__ __launch_bounds__(256)
void dtproj_kernel(const float* __restrict__ xdbl, float* __restrict__ dt,
                   const float* __restrict__ w, const float* __restrict__ bias)
{
    long s = blockIdx.z;
    int idx = blockIdx.x * blockDim.x + threadIdx.x;
    if (idx >= 256 * LSEQ) return;
    int d = idx / LSEQ, l = idx - d * LSEQ;
    const float* xd = xdbl + s * 69120;
    float acc = bias[d];
#pragma unroll
    for (int r = 0; r < 8; ++r) acc += w[d * 8 + r] * xd[(long)r * LSEQ + l];
    dt[s * 442368 + idx] = (acc > 20.f) ? acc : log1pf(__expf(acc));
}

// Selective scan: 256 threads = 256 channels, h[16] in registers, L sequential.
// B_t / C_t broadcast through LDS.  xdbl rows: 0..7 dt_in, 8..23 B, 24..39 C.
__global__ __launch_bounds__(256)
void scan_kernel(const float* __restrict__ dt, const float* __restrict__ xm2,
                 const float* __restrict__ xdbl, const float* __restrict__ A_log,
                 float* __restrict__ y)
{
    long s = blockIdx.x;
    int d = threadIdx.x;
    float a[16], h[16];
#pragma unroll
    for (int n = 0; n < 16; ++n) { a[n] = -__expf(A_log[d * 16 + n]); h[n] = 0.f; }
    const float* dts = dt  + s * 442368 + (long)d * LSEQ;
    const float* xs  = xm2 + s * 442368 + (long)d * LSEQ;
    const float* xd  = xdbl + s * 69120;
    float*       ys  = y   + s * 442368 + (long)d * LSEQ;
    __shared__ float sB[16], sC[16];
    for (int l = 0; l < LSEQ; ++l) {
        if (d < 16)      sB[d]      = xd[(long)(8  + d)      * LSEQ + l];
        else if (d < 32) sC[d - 16] = xd[(long)(24 + d - 16) * LSEQ + l];
        __syncthreads();
        float dtv = dts[l], xv = xs[l], dx = dtv * xv, yv = 0.f;
#pragma unroll
        for (int n = 0; n < 16; ++n) {
            h[n] = __expf(dtv * a[n]) * h[n] + dx * sB[n];
            yv  += h[n] * sC[n];
        }
        ys[l] = yv;
        __syncthreads();
    }
}

// g = (y + xm2*D) * silu(z)  (in place into y).  z = rows 256..511 of xz.
__global__ __launch_bounds__(256)
void gate_kernel(float* __restrict__ y, const float* __restrict__ xm2,
                 const float* __restrict__ xz, const float* __restrict__ Dp)
{
    long s = blockIdx.z;
    int idx = blockIdx.x * blockDim.x + threadIdx.x;
    if (idx >= 256 * LSEQ) return;
    int d = idx / LSEQ, l = idx - d * LSEQ;
    float zv  = xz[s * 884736 + (long)(256 + d) * LSEQ + l];
    float val = y[s * 442368 + idx] + xm2[s * 442368 + idx] * Dp[d];
    y[s * 442368 + idx] = val * silu_f(zv);
}

// out = relu(t2 + mamba_out + identity)   (all [128][1728])
__global__ __launch_bounds__(256)
void addrelu3(const float* __restrict__ t2, const float* __restrict__ mo,
              const float* __restrict__ idp, float* __restrict__ out)
{
    long s = blockIdx.z;
    int idx = blockIdx.x * blockDim.x + threadIdx.x;
    if (idx >= 128 * LSEQ) return;
    long o = s * 221184 + idx;
    out[o] = fmaxf(t2[o] + mo[o] + idp[o], 0.f);
}

// Global average pool over 12^3 -> feats[s][128]
__global__ __launch_bounds__(128)
void pool_kernel(const float* __restrict__ blk, float* __restrict__ feats)
{
    long s = blockIdx.x;
    int c = threadIdx.x;
    const float* p = blk + s * 221184 + (long)c * LSEQ;
    float acc = 0.f;
    for (int l = 0; l < LSEQ; ++l) acc += p[l];
    feats[s * 128 + c] = acc * (1.f / 1728.f);
}

// Final FC: out[b] = concat(feats[b],...) . fc_w + fc_b
__global__ __launch_bounds__(128)
void fc_kernel(const float* __restrict__ feats, const float* __restrict__ fw,
               const float* __restrict__ fb, float* __restrict__ out)
{
    int b = blockIdx.x, t = threadIdx.x;
    __shared__ float red[128];
    float acc = 0.f;
    for (int j = t; j < 896; j += 128) {
        int slot = j >> 7, c = j & 127;
        int s = (slot == 0) ? b : (2 + b * 6 + (slot - 1));
        acc += feats[s * 128 + c] * fw[j];
    }
    red[t] = acc;
    __syncthreads();
    for (int o = 64; o > 0; o >>= 1) { if (t < o) red[t] += red[t + o]; __syncthreads(); }
    if (t == 0) out[b] = red[0] + fb[0];
}

// ---------------------------------------------------------------------------
// Host side
// ---------------------------------------------------------------------------
struct BranchP {
    const float *cw,*cb,*cbn_g,*cbn_b,*cbn_m,*cbn_v;
    const float *b1_w1,*b1_g1,*b1_b1,*b1_m1,*b1_v1;
    const float *b1_w2,*b1_g2,*b1_b2,*b1_m2,*b1_v2;
    const float *b1_dw,*b1_dg,*b1_db,*b1_dm,*b1_dv;
    const float *b2_w1,*b2_g1,*b2_b1,*b2_m1,*b2_v1;
    const float *b2_w2,*b2_g2,*b2_b2,*b2_m2,*b2_v2;
    const float *nin_w,*mbn_g,*mbn_b,*mbn_m,*mbn_v;
    const float *in_proj,*conv_w,*conv_b,*x_proj,*dt_proj,*dt_bias,*A_log,*Dp,*out_proj;
};

static BranchP load_branch(void* const* d_in, int& i) {
    auto nx = [&]() { return (const float*)d_in[i++]; };
    BranchP P;
    P.cw=nx(); P.cb=nx(); P.cbn_g=nx(); P.cbn_b=nx(); P.cbn_m=nx(); P.cbn_v=nx();
    P.b1_w1=nx(); P.b1_g1=nx(); P.b1_b1=nx(); P.b1_m1=nx(); P.b1_v1=nx();
    P.b1_w2=nx(); P.b1_g2=nx(); P.b1_b2=nx(); P.b1_m2=nx(); P.b1_v2=nx();
    P.b1_dw=nx(); P.b1_dg=nx(); P.b1_db=nx(); P.b1_dm=nx(); P.b1_dv=nx();
    P.b2_w1=nx(); P.b2_g1=nx(); P.b2_b1=nx(); P.b2_m1=nx(); P.b2_v1=nx();
    P.b2_w2=nx(); P.b2_g2=nx(); P.b2_b2=nx(); P.b2_m2=nx(); P.b2_v2=nx();
    P.nin_w=nx(); P.mbn_g=nx(); P.mbn_b=nx(); P.mbn_m=nx(); P.mbn_v=nx();
    P.in_proj=nx(); P.conv_w=nx(); P.conv_b=nx(); P.x_proj=nx();
    P.dt_proj=nx(); P.dt_bias=nx(); P.A_log=nx(); P.Dp=nx(); P.out_proj=nx();
    return P;
}

extern "C" void kernel_launch(void* const* d_in, const int* in_sizes, int n_in,
                              void* d_out, int out_size, void* d_ws, size_t ws_size,
                              hipStream_t stream)
{
    (void)in_sizes; (void)n_in; (void)out_size; (void)ws_size;
    int i = 0;
    const float* X = (const float*)d_in[i++];
    BranchP Pb0 = load_branch(d_in, i);
    BranchP Psh = load_branch(d_in, i);
    const float* fc_w = (const float*)d_in[i++];
    const float* fc_b = (const float*)d_in[i++];

    // Workspace (floats), cursor-allocated.  Per-sample strides:
    const size_t SA = 884736, SXZ = 884736, SC = 221184, SH = 442368, SD = 69120;
    float* WS = (float*)d_ws;
    size_t off = 0;
    auto alloc = [&](size_t n) { float* p = WS + off; off += n; return p; };
    float* bufA   = alloc(14 * SA);   // stem out [64][24^3]
    float* bufxz  = alloc(14 * SXZ);  // in_proj out [512][L]
    float* buf1   = alloc(14 * SC);
    float* buf2   = alloc(14 * SC);   // t2 (conv w2 + bn2)
    float* bufx1  = alloc(14 * SC);   // mamba x1
    float* bufmo  = alloc(14 * SC);   // mamba out (+x1)
    float* bufid  = alloc(14 * SC);   // block1 identity
    float* bufblk = alloc(14 * SC);   // block1 output
    float* bufxm2 = alloc(14 * SH);   // silu(conv1d(xm))
    float* bufdt  = alloc(14 * SH);   // softplus dt
    float* bufy   = alloc(14 * SH);   // scan y / gated g
    float* bufxd  = alloc(14 * SD);   // x_proj out [40][L]
    float* feats  = alloc(14 * 128);

    auto run = [&](const BranchP& P, int s0, int ns) {
        float* A0  = bufA   + (size_t)s0 * SA;
        float* XZ  = bufxz  + (size_t)s0 * SXZ;
        float* B1  = buf1   + (size_t)s0 * SC;
        float* B2  = buf2   + (size_t)s0 * SC;
        float* X1  = bufx1  + (size_t)s0 * SC;
        float* MO  = bufmo  + (size_t)s0 * SC;
        float* ID  = bufid  + (size_t)s0 * SC;
        float* BLK = bufblk + (size_t)s0 * SC;
        float* XM2 = bufxm2 + (size_t)s0 * SH;
        float* DT  = bufdt  + (size_t)s0 * SH;
        float* Y   = bufy   + (size_t)s0 * SH;
        float* XD  = bufxd  + (size_t)s0 * SD;
        float* FT  = feats  + (size_t)s0 * 128;

        auto conv = [&](int KS, const float* in, long inStr, const float* W,
                        float* out, long outStr, int Cout, int Cin, int S, int T,
                        int stride, const float* g, const float* b, const float* m,
                        const float* v, const float* res, long resStr, int relu) {
            int N = T * T * T;
            int tiles = ((Cout + 63) / 64) * ((N + 63) / 64);
            dim3 grid(tiles, 1, ns);
            if (KS == 3)
                wmma_conv<3><<<grid, 128, 0, stream>>>(in, W, out, Cout, Cin, S, T,
                    stride, g, b, m, v, res, relu, inStr, outStr, resStr);
            else
                wmma_conv<1><<<grid, 128, 0, stream>>>(in, W, out, Cout, Cin, S, T,
                    stride, g, b, m, v, res, relu, inStr, outStr, resStr);
        };

        dim3 gE256((256 * LSEQ + 255) / 256, 1, ns);   // 256-ch elementwise
        dim3 gE128((128 * LSEQ + 255) / 256, 1, ns);   // 128-ch elementwise

        auto mamba = [&](const float* t2in) {
            // x1 = relu(bn(nin_w @ t2))
            conv(1, t2in, SC, P.nin_w, X1, SC, 128, 128, 12, 12, 1,
                 P.mbn_g, P.mbn_b, P.mbn_m, P.mbn_v, nullptr, 0, 1);
            // xz[512][L] = in_proj @ x1
            conv(1, X1, SC, P.in_proj, XZ, SXZ, 512, 128, 12, 12, 1,
                 nullptr, nullptr, nullptr, nullptr, nullptr, 0, 0);
            dwconv_silu<<<gE256, 256, 0, stream>>>(XZ, XM2, P.conv_w, P.conv_b);
            // x_dbl[40][L] = x_proj @ xm2
            conv(1, XM2, SH, P.x_proj, XD, SD, 40, 256, 12, 12, 1,
                 nullptr, nullptr, nullptr, nullptr, nullptr, 0, 0);
            dtproj_kernel<<<gE256, 256, 0, stream>>>(XD, DT, P.dt_proj, P.dt_bias);
            scan_kernel<<<dim3(ns), 256, 0, stream>>>(DT, XM2, XD, P.A_log, Y);
            gate_kernel<<<gE256, 256, 0, stream>>>(Y, XM2, XZ, P.Dp);
            // mo = out_proj @ g + x1
            conv(1, Y, SH, P.out_proj, MO, SC, 128, 256, 12, 12, 1,
                 nullptr, nullptr, nullptr, nullptr, X1, SC, 0);
        };

        // Stem
        dim3 gStem((64 * 13824 + 255) / 256, 1, ns);
        stem_kernel<<<gStem, 256, 0, stream>>>(X, A0, P.cw, P.cb,
            P.cbn_g, P.cbn_b, P.cbn_m, P.cbn_v, s0);

        // BasicBlock 1 (64->128, stride 2, downsample)
        conv(3, A0, SA, P.b1_w1, B1, SC, 128, 64, 24, 12, 2,
             P.b1_g1, P.b1_b1, P.b1_m1, P.b1_v1, nullptr, 0, 1);
        conv(3, B1, SC, P.b1_w2, B2, SC, 128, 128, 12, 12, 1,
             P.b1_g2, P.b1_b2, P.b1_m2, P.b1_v2, nullptr, 0, 0);
        conv(1, A0, SA, P.b1_dw, ID, SC, 128, 64, 24, 12, 2,
             P.b1_dg, P.b1_db, P.b1_dm, P.b1_dv, nullptr, 0, 0);
        mamba(B2);
        addrelu3<<<gE128, 256, 0, stream>>>(B2, MO, ID, BLK);

        // BasicBlock 2 (128->128, stride 1, identity = input)
        conv(3, BLK, SC, P.b2_w1, B1, SC, 128, 128, 12, 12, 1,
             P.b2_g1, P.b2_b1, P.b2_m1, P.b2_v1, nullptr, 0, 1);
        conv(3, B1, SC, P.b2_w2, B2, SC, 128, 128, 12, 12, 1,
             P.b2_g2, P.b2_b2, P.b2_m2, P.b2_v2, nullptr, 0, 0);
        mamba(B2);
        addrelu3<<<gE128, 256, 0, stream>>>(B2, MO, BLK, B1);

        pool_kernel<<<dim3(ns), 128, 0, stream>>>(B1, FT);
    };

    run(Pb0, 0, 2);    // b0 branch on x[:, 0]
    run(Psh, 2, 12);   // shared branch on x[:, 1:7]

    fc_kernel<<<dim3(2), 128, 0, stream>>>(feats, fc_w, fc_b, (float*)d_out);
}
